// HeteroGNN_71399536329138
// MI455X (gfx1250) — matrix operand order repaired and checked by reading, not compiled, not run on hardware
//
#include <hip/hip_runtime.h>
#include <hip/hip_bf16.h>

// Problem constants (match reference)
#define NCC   200000   // clients
#define NAA   1000     // aggregators
#define EE    1000000  // edges per direction
#define DAA   1024
#define DBB   256
#define FF    128
#define AINN  64
#define HH    128

typedef __attribute__((ext_vector_type(16))) __bf16 v16bf;
typedef __attribute__((ext_vector_type(8)))  float  v8f;

__device__ __forceinline__ float leakyf(float x) { return x >= 0.f ? x : 0.1f * x; }

// ---- CDNA5 async global->LDS staging (ASYNCcnt-tracked DMA, ISA 08_async_tensor §4) ----
// Stage 8KB: 256 threads x 32B each (two b128 async-to-LDS instructions per wave).
__device__ __forceinline__ void stage8k(const __bf16* __restrict__ g, unsigned lds_base, int tid) {
    unsigned long long ga = (unsigned long long)(uintptr_t)g + (unsigned)(tid * 16);
    unsigned la = lds_base + (unsigned)(tid * 16);
    unsigned long long ga2 = ga + 4096u;
    unsigned la2 = la + 4096u;
    asm volatile("global_load_async_to_lds_b128 %0, %1, off" :: "v"(la),  "v"(ga)  : "memory");
    asm volatile("global_load_async_to_lds_b128 %0, %1, off" :: "v"(la2), "v"(ga2) : "memory");
}
__device__ __forceinline__ void wait_async_le2() { asm volatile("s_wait_asynccnt 0x2" ::: "memory"); }
__device__ __forceinline__ void wait_async_0()   { asm volatile("s_wait_asynccnt 0x0" ::: "memory"); }

// Build a WMMA A-fragment (16x32 bf16) from row-major f32.
// 16-bit A layout: lanes 0-15 hold K = kk+0..7 (elts 0-7) and kk+16..23 (elts 8-15);
// lanes 16-31 hold K = kk+8..15 and kk+24..31.  ap = &A[row*lda + kk + half*8].
__device__ __forceinline__ v16bf load_a_frag(const float* __restrict__ ap) {
    float4 f0 = *(const float4*)(ap + 0);
    float4 f1 = *(const float4*)(ap + 4);
    float4 f2 = *(const float4*)(ap + 16);
    float4 f3 = *(const float4*)(ap + 20);
    v16bf a;
    a[0] = (__bf16)f0.x;  a[1] = (__bf16)f0.y;  a[2] = (__bf16)f0.z;  a[3] = (__bf16)f0.w;
    a[4] = (__bf16)f1.x;  a[5] = (__bf16)f1.y;  a[6] = (__bf16)f1.z;  a[7] = (__bf16)f1.w;
    a[8] = (__bf16)f2.x;  a[9] = (__bf16)f2.y;  a[10] = (__bf16)f2.z; a[11] = (__bf16)f2.w;
    a[12] = (__bf16)f3.x; a[13] = (__bf16)f3.y; a[14] = (__bf16)f3.z; a[15] = (__bf16)f3.w;
    return a;
}

// Pack f32 W[K x 128] row-major into WMMA B-fragment layout:
// per (kt, nt) tile: 32 lanes x 16 contiguous bf16 (lane = ((kin/8)%2)*16 + n%16, elt j = kin%8 + (kin/16)*8).
__global__ void pack_b_kernel(const float* __restrict__ W, __bf16* __restrict__ dst, int K) {
    int idx = blockIdx.x * 256 + threadIdx.x;
    if (idx >= K * 128) return;
    int k = idx >> 7;
    int n = idx & 127;
    int kt = k >> 5;
    int kin = k & 31;
    int nt = n >> 4;
    int lane = (((kin >> 3) & 1) << 4) | (n & 15);
    int j = (kin & 7) | ((kin >> 4) << 3);
    dst[((((size_t)kt * 8 + nt) * 32 + lane) << 4) + j] = (__bf16)W[(size_t)k * 128 + n];
}

// Encoder: clients = 0.5*(feat_a@Wa + ba + feat_b@Wb + bb), K concatenated (1024 + 256).
// B tiles double-buffered through LDS via async-to-LDS DMA overlapping the WMMA work.
__global__ __launch_bounds__(256) void enc_gemm_kernel(
    const float* __restrict__ fa, const float* __restrict__ fb,
    const __bf16* __restrict__ Bp,
    const float* __restrict__ ba, const float* __restrict__ bb,
    float* __restrict__ out) {
    __shared__ __align__(64) __bf16 sB[8192];   // 2 x 8KB double buffer
    const int tid = threadIdx.x;
    const int lane = tid & 31;
    const int wave = tid >> 5;
    const int tileRow = blockIdx.x * 128 + wave * 16;
    const bool valid = tileRow < NCC;            // wave-uniform; all waves stay for barriers
    const int half = lane >> 4;
    const int r16 = lane & 15;
    const size_t arow = valid ? (size_t)(tileRow + r16) : 0;  // clamped row for loads
    const unsigned lds_base = (unsigned)(uintptr_t)&sB[0];

    stage8k(Bp, lds_base, tid);                  // kt = 0 -> buffer 0

    v8f acc[8] = {};
#pragma unroll 1
    for (int kt = 0; kt < 40; ++kt) {
        if (kt + 1 < 40) {
            stage8k(Bp + (size_t)(kt + 1) * 4096, lds_base + (unsigned)(((kt + 1) & 1) * 8192), tid);
            wait_async_le2();                    // in-order ASYNCcnt: stage(kt) complete
        } else {
            wait_async_0();
        }
        __syncthreads();                         // staged data visible to all waves

        const float* A; size_t lda; int kk;
        if (kt < 32) { A = fa; lda = DAA; kk = kt * 32; }
        else         { A = fb; lda = DBB; kk = (kt - 32) * 32; }
        v16bf a = load_a_frag(A + arow * lda + kk + half * 8);
        const __bf16* bp = sB + (kt & 1) * 4096 + lane * 16;
        v16bf b[8];
#pragma unroll
        for (int nt = 0; nt < 8; ++nt)           // one ds_load clause, one dscnt wait
            b[nt] = *(const v16bf*)(bp + nt * 512);
#pragma unroll
        for (int nt = 0; nt < 8; ++nt)           // back-to-back WMMA burst
            acc[nt] = __builtin_amdgcn_wmma_f32_16x16x32_bf16(
                false, a, false, b[nt], (short)0, acc[nt], false, false);
        __syncthreads();                         // done reading before next stage overwrites
    }
    if (valid) {
#pragma unroll
        for (int nt = 0; nt < 8; ++nt) {
            const int n = nt * 16 + r16;
            const float bias = 0.5f * (ba[n] + bb[n]);
#pragma unroll
            for (int r = 0; r < 8; ++r) {
                const size_t m = (size_t)tileRow + half * 8 + r;
                out[m * 128 + n] = 0.5f * acc[nt][r] + bias;
            }
        }
    }
}

// Client-side SAGE layer: out = leaky(X@W + bias + S[row]/max(cnt[row],1)), K = 128.
// Entire packed 32KB weight staged to LDS once per block via async DMA.
__global__ __launch_bounds__(256) void sage_gemm_kernel(
    const float* __restrict__ X, const __bf16* __restrict__ Bp,
    const float* __restrict__ S, const int* __restrict__ cnt,
    const float* __restrict__ bias, float* __restrict__ out) {
    __shared__ __align__(64) __bf16 sB[16384];  // 32KB: full K=128 packed weight
    const int tid = threadIdx.x;
    const int lane = tid & 31;
    const int wave = tid >> 5;
    const int tileRow = blockIdx.x * 128 + wave * 16;
    const bool valid = tileRow < NCC;
    const int half = lane >> 4;
    const int r16 = lane & 15;
    const size_t arow = valid ? (size_t)(tileRow + r16) : 0;
    const unsigned lds_base = (unsigned)(uintptr_t)&sB[0];

#pragma unroll
    for (int s = 0; s < 4; ++s)
        stage8k(Bp + (size_t)s * 4096, lds_base + (unsigned)(s * 8192), tid);
    wait_async_0();
    __syncthreads();

    v8f acc[8] = {};
#pragma unroll
    for (int kt = 0; kt < 4; ++kt) {
        v16bf a = load_a_frag(X + arow * 128 + kt * 32 + half * 8);
        const __bf16* bp = sB + kt * 4096 + lane * 16;
        v16bf b[8];
#pragma unroll
        for (int nt = 0; nt < 8; ++nt)
            b[nt] = *(const v16bf*)(bp + nt * 512);
#pragma unroll
        for (int nt = 0; nt < 8; ++nt)
            acc[nt] = __builtin_amdgcn_wmma_f32_16x16x32_bf16(
                false, a, false, b[nt], (short)0, acc[nt], false, false);
    }
    if (valid) {
        float inv[8];
#pragma unroll
        for (int r = 0; r < 8; ++r) {
            int m = tileRow + half * 8 + r;
            int c = cnt[m];
            inv[r] = 1.0f / (float)(c > 1 ? c : 1);
        }
#pragma unroll
        for (int nt = 0; nt < 8; ++nt) {
            const int n = nt * 16 + r16;
            const float bn = bias[n];
#pragma unroll
            for (int r = 0; r < 8; ++r) {
                const size_t m = (size_t)tileRow + half * 8 + r;
                float v = acc[nt][r] + bn + S[m * 128 + n] * inv[r];
                out[m * 128 + n] = leakyf(v);
            }
        }
    }
}

// Per-destination in-degree counts for both edge directions.
__global__ void count_kernel(const int* __restrict__ dst_c2a, const int* __restrict__ dst_a2c,
                             int* __restrict__ cnt_c2a, int* __restrict__ cnt_a2c) {
    int e = blockIdx.x * 256 + threadIdx.x;
    if (e >= EE) return;
    atomicAdd(&cnt_c2a[dst_c2a[e]], 1);
    atomicAdd(&cnt_a2c[dst_a2c[e]], 1);
}

// Serial exclusive scan over NAA=1000 counts (tiny).
__global__ void scan_kernel(const int* __restrict__ cnt, int* __restrict__ offs) {
    if (threadIdx.x == 0 && blockIdx.x == 0) {
        int s = 0;
        for (int i = 0; i < NAA; ++i) { offs[i] = s; s += cnt[i]; }
        offs[NAA] = s;
    }
}

__global__ void csr_fill_kernel(const int* __restrict__ src, const int* __restrict__ dst,
                                const int* __restrict__ offs, int* __restrict__ cursor,
                                int* __restrict__ col) {
    int e = blockIdx.x * 256 + threadIdx.x;
    if (e >= EE) return;
    int d = dst[e];
    int p = atomicAdd(&cursor[d], 1);
    col[offs[d] + p] = src[e];
}

// c2a aggregation: A[a] = mean over edge list of X[col[j]] (contention-free, L2-resident X).
__global__ __launch_bounds__(128) void aggr_gather_kernel(
    const float* __restrict__ X, const int* __restrict__ offs,
    const int* __restrict__ col, float* __restrict__ A) {
    int a = blockIdx.x;
    int f = threadIdx.x;
    int s = offs[a], e = offs[a + 1];
    float acc = 0.f;
    for (int j = s; j < e; ++j) acc += X[(size_t)col[j] * 128 + f];
    int c = e - s;
    A[(size_t)a * 128 + f] = acc / (float)(c > 1 ? c : 1);
}

// agg1 = leaky(A1@W1cl + b1cl + x_agg@W1cr)  (NA=1000 rows -> plain VALU is fine)
__global__ void agg1_kernel(const float* __restrict__ A1, const float* __restrict__ Xagg,
                            const float* __restrict__ W1cl, const float* __restrict__ b1cl,
                            const float* __restrict__ W1cr, float* __restrict__ agg1) {
    int idx = blockIdx.x * 256 + threadIdx.x;
    if (idx >= NAA * 128) return;
    int a = idx >> 7, n = idx & 127;
    float acc = b1cl[n];
    for (int k = 0; k < 128; ++k) acc += A1[(size_t)a * 128 + k] * W1cl[(size_t)k * 128 + n];
    for (int k = 0; k < AINN; ++k) acc += Xagg[(size_t)a * AINN + k] * W1cr[(size_t)k * 128 + n];
    agg1[idx] = leakyf(acc);
}

// Y = X[NAA x K] @ W[K x 128]  (tiny)
__global__ void small_gemm_kernel(const float* __restrict__ X, const float* __restrict__ W,
                                  float* __restrict__ Y, int K) {
    int idx = blockIdx.x * 256 + threadIdx.x;
    if (idx >= NAA * 128) return;
    int a = idx >> 7, n = idx & 127;
    float acc = 0.f;
    for (int k = 0; k < K; ++k) acc += X[(size_t)a * K + k] * W[(size_t)k * 128 + n];
    Y[idx] = acc;
}

// a2c scatter: S[dst] += Y[src]  (linearized aggregation; Y has only 1000 rows, L2-hot)
__global__ void scatter_kernel(const float* __restrict__ Y, const int* __restrict__ src,
                               const int* __restrict__ dst, float* __restrict__ S) {
    int t = blockIdx.x * 256 + threadIdx.x;
    int e = t >> 5;
    if (e >= EE) return;
    int q = (t & 31) * 4;
    int s = src[e], d = dst[e];
    float4 v = *(const float4*)(Y + (size_t)s * 128 + q);
    float* o = S + (size_t)d * 128 + q;
    atomicAdd(o + 0, v.x);
    atomicAdd(o + 1, v.y);
    atomicAdd(o + 2, v.z);
    atomicAdd(o + 3, v.w);
}

// out[r] = dot(cli2[r], Wout) + bout
__global__ void out_kernel(const float* __restrict__ cli2, const float* __restrict__ Wout,
                           const float* __restrict__ bout, float* __restrict__ out) {
    int r = blockIdx.x * 256 + threadIdx.x;
    if (r >= NCC) return;
    float acc = bout[0];
    const float* x = cli2 + (size_t)r * 128;
    for (int k = 0; k < 128; k += 4) {
        float4 a = *(const float4*)(x + k);
        float4 w = *(const float4*)(Wout + k);
        acc += a.x * w.x + a.y * w.y + a.z * w.z + a.w * w.w;
    }
    out[r] = acc;
}

extern "C" void kernel_launch(void* const* d_in, const int* in_sizes, int n_in,
                              void* d_out, int out_size, void* d_ws, size_t ws_size,
                              hipStream_t stream) {
    (void)in_sizes; (void)n_in; (void)out_size; (void)ws_size;
    const float* feat_a   = (const float*)d_in[0];
    const float* feat_b   = (const float*)d_in[1];
    const float* x_agg    = (const float*)d_in[2];
    const int*   e_c2a_src = (const int*)d_in[3];
    const int*   e_c2a_dst = (const int*)d_in[4];
    const int*   e_a2c_src = (const int*)d_in[5];
    const int*   e_a2c_dst = (const int*)d_in[6];
    const float* Wa   = (const float*)d_in[7];
    const float* ba   = (const float*)d_in[8];
    const float* Wb   = (const float*)d_in[9];
    const float* bb   = (const float*)d_in[10];
    const float* W1cl = (const float*)d_in[11];
    const float* b1cl = (const float*)d_in[12];
    const float* W1cr = (const float*)d_in[13];
    const float* W1al = (const float*)d_in[14];
    const float* b1al = (const float*)d_in[15];
    const float* W1ar = (const float*)d_in[16];
    // d_in[17..19] = W2cl, b2cl, W2cr: agg2 is dead code (output uses only cli2)
    const float* W2al = (const float*)d_in[20];
    const float* b2al = (const float*)d_in[21];
    const float* W2ar = (const float*)d_in[22];
    const float* Wout = (const float*)d_in[23];
    const float* bout = (const float*)d_in[24];
    float* outp = (float*)d_out;

    // Workspace carving (~315 MB total)
    char* p = (char*)d_ws;
    auto carve = [&](size_t bytes) { char* r = p; p += (bytes + 255) & ~(size_t)255; return r; };
    float* clients = (float*)carve((size_t)NCC * 128 * 4);
    float* cli1    = (float*)carve((size_t)NCC * 128 * 4);
    float* S       = (float*)carve((size_t)NCC * 128 * 4);   // reused for S1 then S2
    float* A1      = (float*)carve((size_t)NAA * 128 * 4);
    float* agg1    = (float*)carve((size_t)NAA * 128 * 4);
    float* Y1      = (float*)carve((size_t)NAA * 128 * 4);
    float* Y2      = (float*)carve((size_t)NAA * 128 * 4);
    __bf16* WencP  = (__bf16*)carve((size_t)(DAA + DBB) * 128 * 2);
    __bf16* W1arP  = (__bf16*)carve((size_t)128 * 128 * 2);
    __bf16* W2arP  = (__bf16*)carve((size_t)128 * 128 * 2);
    int* cnt_c2a   = (int*)carve((size_t)NAA * 4);
    int* cnt_a2c   = (int*)carve((size_t)NCC * 4);
    int* offs      = (int*)carve((size_t)(NAA + 1) * 4);
    int* cursor    = (int*)carve((size_t)NAA * 4);
    int* col       = (int*)carve((size_t)EE * 4);
    float* cli2 = clients;  // clients is fully consumed before cli2 is produced

    hipMemsetAsync(cnt_c2a, 0, (size_t)NAA * 4, stream);
    hipMemsetAsync(cnt_a2c, 0, (size_t)NCC * 4, stream);
    hipMemsetAsync(cursor, 0, (size_t)NAA * 4, stream);
    hipMemsetAsync(S, 0, (size_t)NCC * 128 * 4, stream);

    // Pack weights into WMMA B-fragment layout (Wa as K-tiles 0..31, Wb as 32..39)
    pack_b_kernel<<<(DAA * 128 + 255) / 256, 256, 0, stream>>>(Wa, WencP, DAA);
    pack_b_kernel<<<(DBB * 128 + 255) / 256, 256, 0, stream>>>(Wb, WencP + (size_t)DAA * 128, DBB);
    pack_b_kernel<<<(128 * 128 + 255) / 256, 256, 0, stream>>>(W1ar, W1arP, 128);
    pack_b_kernel<<<(128 * 128 + 255) / 256, 256, 0, stream>>>(W2ar, W2arP, 128);

    // Graph preprocessing
    count_kernel<<<(EE + 255) / 256, 256, 0, stream>>>(e_c2a_dst, e_a2c_dst, cnt_c2a, cnt_a2c);
    scan_kernel<<<1, 32, 0, stream>>>(cnt_c2a, offs);
    csr_fill_kernel<<<(EE + 255) / 256, 256, 0, stream>>>(e_c2a_src, e_c2a_dst, offs, cursor, col);

    // Encoder (HBM-bound: ~1.02 GB feature read)
    enc_gemm_kernel<<<(NCC + 127) / 128, 256, 0, stream>>>(feat_a, feat_b, WencP, ba, bb, clients);

    // Layer 1, client side: cli1 = leaky(clients@W1ar + S1/cnt + b1al), S1 = scatter of x_agg@W1al
    small_gemm_kernel<<<(NAA * 128 + 255) / 256, 256, 0, stream>>>(x_agg, W1al, Y1, AINN);
    scatter_kernel<<<(EE * 32 + 255) / 256, 256, 0, stream>>>(Y1, e_a2c_src, e_a2c_dst, S);
    sage_gemm_kernel<<<(NCC + 127) / 128, 256, 0, stream>>>(clients, W1arP, S, cnt_a2c, b1al, cli1);

    // Layer 1, aggregator side: agg1 = leaky(mean_c2a(clients)@W1cl + b1cl + x_agg@W1cr)
    aggr_gather_kernel<<<NAA, 128, 0, stream>>>(clients, offs, col, A1);
    agg1_kernel<<<(NAA * 128 + 255) / 256, 256, 0, stream>>>(A1, x_agg, W1cl, b1cl, W1cr, agg1);

    // Layer 2, client side: cli2 = leaky(cli1@W2ar + S2/cnt + b2al), S2 = scatter of agg1@W2al
    small_gemm_kernel<<<(NAA * 128 + 255) / 256, 256, 0, stream>>>(agg1, W2al, Y2, HH);
    hipMemsetAsync(S, 0, (size_t)NCC * 128 * 4, stream);
    scatter_kernel<<<(EE * 32 + 255) / 256, 256, 0, stream>>>(Y2, e_a2c_src, e_a2c_dst, S);
    sage_gemm_kernel<<<(NCC + 127) / 128, 256, 0, stream>>>(cli1, W2arP, S, cnt_a2c, b2al, cli2);

    // Head
    out_kernel<<<(NCC + 255) / 256, 256, 0, stream>>>(cli2, Wout, bout, outp);
}
